// QLinearA8W8_86552180949216
// MI455X (gfx1250) — compile-verified, matrix operand order checked by reading, not run on hardware
//
#include <hip/hip_runtime.h>
#include <stdint.h>

#define MDIM 8192
#define KDIM 4096
#define NDIM 4096

typedef int v8i __attribute__((ext_vector_type(8)));

// Monotonic order-preserving mapping float -> uint32 so we can atomicMax on uints.
__device__ __forceinline__ uint32_t fkey(float f) {
    uint32_t b = __float_as_uint(f);
    return (b & 0x80000000u) ? ~b : (b | 0x80000000u);
}
__device__ __forceinline__ float funkey(uint32_t u) {
    uint32_t b = (u & 0x80000000u) ? (u ^ 0x80000000u) : ~u;
    return __uint_as_float(b);
}

__global__ void qgemm_init_kernel(uint32_t* key) {
    if (threadIdx.x == 0) *key = 0u;  // identity for max in key space
}

__global__ void qgemm_max_kernel(const float4* __restrict__ x, uint32_t* key, int n4) {
    __shared__ float smax[256];
    float m = -3.402823466e38f;
    for (int i = blockIdx.x * blockDim.x + threadIdx.x; i < n4;
         i += gridDim.x * blockDim.x) {
        float4 v = x[i];
        m = fmaxf(m, fmaxf(fmaxf(v.x, v.y), fmaxf(v.z, v.w)));
    }
    smax[threadIdx.x] = m;
    __syncthreads();
    for (int s = 128; s > 0; s >>= 1) {
        if (threadIdx.x < s) smax[threadIdx.x] = fmaxf(smax[threadIdx.x], smax[threadIdx.x + s]);
        __syncthreads();
    }
    if (threadIdx.x == 0) atomicMax(key, fkey(smax[0]));
}

__device__ __forceinline__ int q8(float v, float s) {
    int q = (int)rintf(v * s);              // round-half-even, matches jnp.round
    q = q < -128 ? -128 : (q > 127 ? 127 : q);
    return q & 0xFF;
}

__global__ void qgemm_quant_kernel(const float4* __restrict__ x, const uint32_t* __restrict__ key,
                                   float* clampv_out, int32_t* __restrict__ x8, int n4) {
    const float clampv = funkey(*key);
    const float s = 127.0f / clampv;
    if (blockIdx.x == 0 && threadIdx.x == 0) *clampv_out = clampv;
    for (int i = blockIdx.x * blockDim.x + threadIdx.x; i < n4;
         i += gridDim.x * blockDim.x) {
        float4 v = x[i];
        int packed = q8(v.x, s) | (q8(v.y, s) << 8) | (q8(v.z, s) << 16) | (q8(v.w, s) << 24);
        x8[i] = packed;
    }
}

// Each wave: 64x64 output tile = 4x4 WMMA accumulators.
// Block: 256 threads = 8 waves arranged 2 (M) x 4 (N) -> 128 x 256 per block.
__global__ __launch_bounds__(256) void qgemm_wmma_kernel(
    const int8_t* __restrict__ A8,      // [M,K] row-major int8 (quantized x)
    const int8_t* __restrict__ W8,      // [N,K] row-major int8 (B^T: column n of B is row n of W)
    const float* __restrict__ qscales,  // [N]
    const float* __restrict__ biasv,    // [N]
    const float* __restrict__ clampv_p,
    float* __restrict__ out)            // [M,N] f32
{
    const int lane = threadIdx.x & 31;
    const int wave = threadIdx.x >> 5;
    const int wm = wave >> 2;           // 0..1
    const int wn = wave & 3;            // 0..3
    const int m0 = blockIdx.y * 128 + wm * 64;
    const int n0 = blockIdx.x * 256 + wn * 64;

    // 8-bit A/B fragment layout (ISA 7.12.2): lane&15 selects the row (A) / col (B),
    // lane>>4 selects the +8 byte half; VGPR pairs hold 8-byte K-chunks at K = {0,16,32,48}+koff.
    const int r    = lane & 15;
    const int koff = (lane >> 4) * 8;

    union Frag { v8i v; uint64_t q[4]; };

    const int8_t* aBase[4];
    const int8_t* bBase[4];
#pragma unroll
    for (int i = 0; i < 4; ++i) aBase[i] = A8 + (size_t)(m0 + 16 * i + r) * KDIM + koff;
#pragma unroll
    for (int j = 0; j < 4; ++j) bBase[j] = W8 + (size_t)(n0 + 16 * j + r) * KDIM + koff;

    v8i acc[4][4] = {};

    for (int kk = 0; kk < KDIM; kk += 64) {
        Frag a[4], b[4];
#pragma unroll
        for (int i = 0; i < 4; ++i) {
            const uint64_t* p = (const uint64_t*)(aBase[i] + kk);
            a[i].q[0] = p[0]; a[i].q[1] = p[2]; a[i].q[2] = p[4]; a[i].q[3] = p[6];
        }
#pragma unroll
        for (int j = 0; j < 4; ++j) {
            const uint64_t* p = (const uint64_t*)(bBase[j] + kk);
            b[j].q[0] = p[0]; b[j].q[1] = p[2]; b[j].q[2] = p[4]; b[j].q[3] = p[6];
        }
#pragma unroll
        for (int i = 0; i < 4; ++i)
#pragma unroll
            for (int j = 0; j < 4; ++j)
                acc[i][j] = __builtin_amdgcn_wmma_i32_16x16x64_iu8(
                    /*sgn_a=*/true, a[i].v, /*sgn_b=*/true, b[j].v,
                    acc[i][j], /*reuse_a=*/false, /*reuse_b=*/false);
    }

    // Epilogue: out = acc * (clampv/127) * qscales[col] + bias[col]
    // C/D layout: VGPR p -> row p + 8*(lane>>4); col = n-tile + (lane&15).
    const float alpha = (*clampv_p) * (1.0f / 127.0f);
    const int hi = lane >> 4;
#pragma unroll
    for (int j = 0; j < 4; ++j) {
        const int col = n0 + 16 * j + r;
        const float sc = qscales[col] * alpha;
        const float bi = biasv[col];
#pragma unroll
        for (int i = 0; i < 4; ++i) {
#pragma unroll
            for (int p = 0; p < 8; ++p) {
                const int row = m0 + 16 * i + p + 8 * hi;
                out[(size_t)row * NDIM + col] = (float)acc[i][j][p] * sc + bi;
            }
        }
    }
}

extern "C" void kernel_launch(void* const* d_in, const int* in_sizes, int n_in,
                              void* d_out, int out_size, void* d_ws, size_t ws_size,
                              hipStream_t stream) {
    const float*  x    = (const float*)d_in[0];
    const int8_t* w8   = (const int8_t*)d_in[1];
    const float*  qs   = (const float*)d_in[2];
    const float*  bias = (const float*)d_in[3];
    float* out = (float*)d_out;

    // Workspace layout: [0..3] uint32 max-key, [8..11] decoded clampv float,
    // [256 ..] 32 MB packed int8 activations.
    uint32_t* key    = (uint32_t*)d_ws;
    float*    clampv = (float*)((char*)d_ws + 8);
    int32_t*  x8     = (int32_t*)((char*)d_ws + 256);

    const int n4 = (MDIM * KDIM) / 4;

    qgemm_init_kernel<<<1, 64, 0, stream>>>(key);
    qgemm_max_kernel<<<1024, 256, 0, stream>>>((const float4*)x, key, n4);
    qgemm_quant_kernel<<<2048, 256, 0, stream>>>((const float4*)x, key, clampv, x8, n4);

    dim3 grid(NDIM / 256, MDIM / 128);
    qgemm_wmma_kernel<<<grid, 256, 0, stream>>>(x8 ? (const int8_t*)x8 : nullptr, w8, qs, bias,
                                                clampv, out);
}
